// CTMBlock_57423712748044
// MI455X (gfx1250) — matrix operand order repaired and checked by reading, not compiled, not run on hardware
//
#include <hip/hip_runtime.h>
#include <hip/hip_bf16.h>

typedef _Float16 v16h __attribute__((ext_vector_type(16)));
typedef _Float16 v8h  __attribute__((ext_vector_type(8)));
typedef float    v8f  __attribute__((ext_vector_type(8)));

#define B_SZ 8
#define T_SZ 2048
#define D_SZ 256

// ---------------------------------------------------------------- utilities

__device__ __forceinline__ v8f wmma16(v16h a, v16h b, v8f c) {
  // D = A(16x32 f16) * B(32x16 f16) + C(16x16 f32)
  return __builtin_amdgcn_wmma_f32_16x16x32_f16(false, a, false, b, (short)0, c,
                                                false, false);
}

// A-fragment (16x32 f16): lane L<16 holds row L, K = {k0..k0+15}
// with k0 = 0; lanes 16..31 hold the same rows with k0 = 16.
__device__ __forceinline__ v16h load_a16(const _Float16* p) {
  v8h x0 = *(const v8h*)p;
  v8h x1 = *(const v8h*)(p + 8);
  v16h a;
#pragma unroll
  for (int i = 0; i < 8; ++i) { a[i] = x0[i]; a[i + 8] = x1[i]; }
  return a;
}

__device__ __forceinline__ float waveSum(float v) {
#pragma unroll
  for (int m = 1; m < 32; m <<= 1) v += __shfl_xor(v, m, 32);
  return v;
}
__device__ __forceinline__ float waveMax(float v) {
#pragma unroll
  for (int m = 1; m < 32; m <<= 1) v = fmaxf(v, __shfl_xor(v, m, 32));
  return v;
}

// ---------------------------------------------------------------- small kernels

__global__ void zero_f(float* p, int n) {
  int i = blockIdx.x * blockDim.x + threadIdx.x;
  if (i < n) p[i] = 0.0f;
}

__global__ void cvt_f16(const float* __restrict__ src, _Float16* __restrict__ dst,
                        long long n) {
  long long stride = (long long)gridDim.x * blockDim.x;
  for (long long i = (long long)blockIdx.x * blockDim.x + threadIdx.x; i < n;
       i += stride)
    dst[i] = (_Float16)src[i];
}

// ---------------------------------------------------------------- WMMA GEMM
// C[M,N] = scale * (A16[M,K] @ W16[N,K]^T) + bias, optional relu.
// Optional outputs: f32, f16, and f16-transposed-per-batch (for V -> [B][D][T]).
// Block: 256 threads = 8 waves; wave computes 16(M) x 64(N); block tile 64x128.
__global__ __launch_bounds__(256)
void gemm_wmma(const _Float16* __restrict__ A, const _Float16* __restrict__ W,
               const float* __restrict__ bias, float* __restrict__ outF,
               _Float16* __restrict__ outH, _Float16* __restrict__ outT, int M,
               int N, int K, float scale, int relu, long long aStride,
               long long wStride, long long oStride) {
  const int lane = threadIdx.x & 31;
  const int wid = threadIdx.x >> 5;
  const int rowBase = blockIdx.x * 64 + (wid & 3) * 16;
  const int colBase = blockIdx.y * 128 + (wid >> 2) * 64;
  const int bz = blockIdx.z;
  const _Float16* Ab = A + (long long)bz * aStride;
  const _Float16* Wb = W + (long long)bz * wStride;

  const int arow = rowBase + (lane & 15);
  const int k0 = (lane >> 4) << 4;   // A-fragment K half within 32-block
  const int kh = (lane >> 4) << 4;   // B-fragment K half within 32-block
  const _Float16* ap = Ab + (long long)arow * K + k0;
  const _Float16* wp0 = Wb + (long long)(colBase + 0  + (lane & 15)) * K + kh;
  const _Float16* wp1 = Wb + (long long)(colBase + 16 + (lane & 15)) * K + kh;
  const _Float16* wp2 = Wb + (long long)(colBase + 32 + (lane & 15)) * K + kh;
  const _Float16* wp3 = Wb + (long long)(colBase + 48 + (lane & 15)) * K + kh;

  v8f acc0 = {}, acc1 = {}, acc2 = {}, acc3 = {};
#pragma unroll 4
  for (int kb = 0; kb < K; kb += 32) {
    __builtin_prefetch(ap + kb + 128, 0, 0);  // global_prefetch_b8
    v16h a = load_a16(ap + kb);
    acc0 = wmma16(a, *(const v16h*)(wp0 + kb), acc0);
    acc1 = wmma16(a, *(const v16h*)(wp1 + kb), acc1);
    acc2 = wmma16(a, *(const v16h*)(wp2 + kb), acc2);
    acc3 = wmma16(a, *(const v16h*)(wp3 + kb), acc3);
  }

  const int rbase = rowBase + ((lane >> 4) << 3);
#pragma unroll
  for (int t = 0; t < 4; ++t) {
    v8f acc = (t == 0) ? acc0 : (t == 1) ? acc1 : (t == 2) ? acc2 : acc3;
    int col = colBase + 16 * t + (lane & 15);
    float bv = bias ? bias[col] : 0.0f;
#pragma unroll
    for (int r = 0; r < 8; ++r) {
      int row = rbase + r;
      float v = acc[r] * scale + bv;
      if (relu) v = fmaxf(v, 0.0f);
      long long oi = (long long)bz * oStride + (long long)row * N + col;
      if (outF) outF[oi] = v;
      if (outH) outH[oi] = (_Float16)v;
      if (outT) {  // per-batch transpose: [B][N][T], tokens-per-batch = T_SZ
        int bb = row >> 11;            // row / T_SZ
        int s = row & (T_SZ - 1);
        outT[((long long)bb * N + col) * T_SZ + s] = (_Float16)v;
      }
    }
  }
}

// ---------------------------------------------------------------- LayerNorm
// One wave per row of 256 f32; writes f16.
__global__ __launch_bounds__(256)
void ln_kernel(const float* __restrict__ x, const float* __restrict__ g,
               const float* __restrict__ bta, _Float16* __restrict__ outH) {
  const int lane = threadIdx.x & 31;
  const int row = blockIdx.x * 8 + (threadIdx.x >> 5);
  const float* xr = x + (long long)row * D_SZ;
  float v[8], s = 0.f, sq = 0.f;
#pragma unroll
  for (int i = 0; i < 8; ++i) {
    v[i] = xr[lane + 32 * i];
    s += v[i];
    sq += v[i] * v[i];
  }
  s = waveSum(s);
  sq = waveSum(sq);
  float mu = s * (1.0f / D_SZ);
  float var = sq * (1.0f / D_SZ) - mu * mu;
  float rstd = rsqrtf(var + 1e-5f);
#pragma unroll
  for (int i = 0; i < 8; ++i) {
    int c = lane + 32 * i;
    outH[(long long)row * D_SZ + c] = (_Float16)((v[i] - mu) * rstd * g[c] + bta[c]);
  }
}

// ------------------------------------------------------- softmax row statistics
// mode 0: plain softmax; row_max(attn) = 1/Z -> atomicAdd into confSum[b].
// mode 1: sharpened by effp[b]; store per-row max (m2) and sum-exp (z2).
__global__ __launch_bounds__(256)
void attn_stats(const _Float16* __restrict__ sim, const float* __restrict__ effp,
                float* __restrict__ confSum, float* __restrict__ m2,
                float* __restrict__ z2, int mode) {
  const int lane = threadIdx.x & 31;
  const long long row = (long long)blockIdx.x * 8 + (threadIdx.x >> 5);
  const int b = (int)(row >> 11);  // row / T_SZ
  const float p = (mode == 0) ? 1.0f : effp[b];
  const _Float16* sr = sim + row * T_SZ;
  float m = -1e30f;
  for (int j = lane; j < T_SZ; j += 32) m = fmaxf(m, p * (float)sr[j]);
  m = waveMax(m);
  float Z = 0.f;
  for (int j = lane; j < T_SZ; j += 32) Z += __expf(p * (float)sr[j] - m);
  Z = waveSum(Z);
  if (lane == 0) {
    if (mode == 0)
      atomicAdd(&confSum[b], 1.0f / Z);
    else {
      m2[row] = m;
      z2[row] = Z;
    }
  }
}

__global__ void conf_kernel(const float* __restrict__ confSum,
                            float* __restrict__ effp, float* __restrict__ effpSum) {
  int t = threadIdx.x;
  if (t < B_SZ) {
    const float uniform = 1.0f / T_SZ;
    float c = (confSum[t] * (1.0f / T_SZ) - uniform) / (1.0f - uniform + 1e-6f);
    c = fminf(fmaxf(c, 0.0f), 1.0f);
    float p = 2.0f + 2.0f * c;  // P_MIN + (P_MAX-P_MIN)*conf
    effp[t] = p;
    atomicAdd(effpSum, p);
  }
}

// ------------------------------------------------- context = softmax_sharp @ V
// One wave per (16 rows x 64 cols) of context for one batch. P is rebuilt
// lane-local in A-fragment layout from the f16 sim buffer; V is consumed from
// the [B][D][T] transposed layout so B-fragments are contiguous-K loads.
__global__ __launch_bounds__(32)
void attn_ctx(const _Float16* __restrict__ sim, const _Float16* __restrict__ VT,
              const float* __restrict__ effp, const float* __restrict__ m2,
              const float* __restrict__ z2, _Float16* __restrict__ ctx) {
  const int lane = threadIdx.x & 31;
  const int rowBase = blockIdx.x * 16;
  const int colBase = blockIdx.y * 64;
  const int b = blockIdx.z;
  const float p = effp[b];
  const int arow = rowBase + (lane & 15);
  const _Float16* sr = sim + ((long long)b * T_SZ + arow) * T_SZ;
  const float m2r = m2[b * T_SZ + arow];
  const int k0 = (lane >> 4) << 4;
  const int kh = (lane >> 4) << 4;
  const _Float16* vp0 = VT + ((long long)b * D_SZ + colBase + 0  + (lane & 15)) * T_SZ + kh;
  const _Float16* vp1 = VT + ((long long)b * D_SZ + colBase + 16 + (lane & 15)) * T_SZ + kh;
  const _Float16* vp2 = VT + ((long long)b * D_SZ + colBase + 32 + (lane & 15)) * T_SZ + kh;
  const _Float16* vp3 = VT + ((long long)b * D_SZ + colBase + 48 + (lane & 15)) * T_SZ + kh;

  v8f acc0 = {}, acc1 = {}, acc2 = {}, acc3 = {};
  for (int sb = 0; sb < T_SZ; sb += 32) {
    v8h x0 = *(const v8h*)(sr + sb + k0);
    v8h x1 = *(const v8h*)(sr + sb + k0 + 8);
    v16h a;
#pragma unroll
    for (int i = 0; i < 8; ++i) {
      a[i]     = (_Float16)__expf(p * (float)x0[i] - m2r);
      a[i + 8] = (_Float16)__expf(p * (float)x1[i] - m2r);
    }
    acc0 = wmma16(a, *(const v16h*)(vp0 + sb), acc0);
    acc1 = wmma16(a, *(const v16h*)(vp1 + sb), acc1);
    acc2 = wmma16(a, *(const v16h*)(vp2 + sb), acc2);
    acc3 = wmma16(a, *(const v16h*)(vp3 + sb), acc3);
  }

  const int rbase = rowBase + ((lane >> 4) << 3);
  float zinv[8];
#pragma unroll
  for (int r = 0; r < 8; ++r) zinv[r] = 1.0f / z2[b * T_SZ + rbase + r];
#pragma unroll
  for (int t = 0; t < 4; ++t) {
    v8f acc = (t == 0) ? acc0 : (t == 1) ? acc1 : (t == 2) ? acc2 : acc3;
    int col = colBase + 16 * t + (lane & 15);
#pragma unroll
    for (int r = 0; r < 8; ++r) {
      int row = rbase + r;
      ctx[((long long)b * T_SZ + row) * D_SZ + col] = (_Float16)(acc[r] * zinv[r]);
    }
  }
}

// ---------------------------------------------- z update + prediction MSE loss
__global__ __launch_bounds__(256)
void update_kernel(const float* __restrict__ z, const float* __restrict__ target,
                   const float* __restrict__ zpred,
                   const float* __restrict__ logTau, float* __restrict__ zOut,
                   _Float16* __restrict__ z16, float* __restrict__ lossSum,
                   long long n) {
  float local = 0.f;
  long long stride = (long long)gridDim.x * blockDim.x;
  for (long long i = (long long)blockIdx.x * blockDim.x + threadIdx.x; i < n;
       i += stride) {
    int d = (int)(i & (D_SZ - 1));
    float tau = __expf(logTau[d]);
    float zi = z[i];
    float zn = zi + (target[i] - zi) / (tau + 1e-6f);  // DT = 1
    zOut[i] = zn;
    z16[i] = (_Float16)zn;
    float df = zpred[i] - zn;
    local += df * df;
  }
  local = waveSum(local);
  if ((threadIdx.x & 31) == 0) atomicAdd(lossSum, local);
}

__global__ void finalize(const float* __restrict__ lossSum,
                         const float* __restrict__ effpSum, float* __restrict__ out,
                         long long btd) {
  out[btd] = lossSum[0] / (3.0f * (float)btd);
  out[btd + 1] = effpSum[0] / (3.0f * (float)B_SZ);
}

// ---------------------------------------------------------------- launcher

extern "C" void kernel_launch(void* const* d_in, const int* in_sizes, int n_in,
                              void* d_out, int out_size, void* d_ws,
                              size_t ws_size, hipStream_t stream) {
  (void)in_sizes; (void)n_in; (void)out_size; (void)ws_size;
  const int B = B_SZ, T = T_SZ, D = D_SZ;
  const long long M = (long long)B * T;  // 16384 tokens
  const long long MD = M * D;            // 4,194,304
  const float inv_sqrt_d = 1.0f / 16.0f;

  const float* zIn = (const float*)d_in[0];
  const float* in_w = (const float*)d_in[1];
  const float* in_b = (const float*)d_in[2];
  const float* ln_g = (const float*)d_in[3];
  const float* ln_b = (const float*)d_in[4];
  const float* q_w = (const float*)d_in[5];
  const float* q_b = (const float*)d_in[6];
  const float* k_w = (const float*)d_in[7];
  const float* k_b = (const float*)d_in[8];
  const float* v_w = (const float*)d_in[9];
  const float* v_b = (const float*)d_in[10];
  const float* o_w = (const float*)d_in[11];
  const float* o_b = (const float*)d_in[12];
  const float* log_tau = (const float*)d_in[13];
  const float* p1_w = (const float*)d_in[14];
  const float* p1_b = (const float*)d_in[15];
  const float* p2_w = (const float*)d_in[16];
  const float* p2_b = (const float*)d_in[17];

  // workspace carve-up (256B aligned slabs)
  char* base = (char*)d_ws;
  size_t off = 0;
  auto alloc = [&](size_t bytes) -> char* {
    char* p = base + off;
    off = (off + bytes + 255) & ~(size_t)255;
    return p;
  };
  _Float16* sim16 = (_Float16*)alloc((size_t)B * T * T * 2);  // 64 MB (L2 resident)
  _Float16* z16 = (_Float16*)alloc(MD * 2);
  _Float16* h16 = (_Float16*)alloc(MD * 2);
  _Float16* q16 = (_Float16*)alloc(MD * 2);
  _Float16* k16 = (_Float16*)alloc(MD * 2);
  _Float16* vt16 = (_Float16*)alloc(MD * 2);  // [B][D][T]
  _Float16* ctx16 = (_Float16*)alloc(MD * 2);
  _Float16* r116 = (_Float16*)alloc(MD * 2);
  float* fbufA = (float*)alloc(MD * 4);  // pre-LN activations, later 'target'
  float* fbufB = (float*)alloc(MD * 4);  // z_pred
  float* zState = (float*)alloc(MD * 4);
  float* m2 = (float*)alloc(M * 4);
  float* z2 = (float*)alloc(M * 4);
  const int DW = D * D;  // 65536
  _Float16* w16[7];
  for (int i = 0; i < 7; ++i) w16[i] = (_Float16*)alloc((size_t)DW * 2);
  float* smallBuf = (float*)alloc(256);
  float* confSum = smallBuf;       // [0..7]
  float* effp = smallBuf + 8;      // [8..15]
  float* lossSum = smallBuf + 16;  // [16]
  float* effpSum = smallBuf + 17;  // [17]

  // one-time per launch: f16 weights, f16 z, zeroed accumulators
  const float* wsrc[7] = {in_w, q_w, k_w, v_w, o_w, p1_w, p2_w};
  for (int i = 0; i < 7; ++i)
    cvt_f16<<<64, 256, 0, stream>>>(wsrc[i], w16[i], (long long)DW);
  cvt_f16<<<2048, 256, 0, stream>>>(zIn, z16, MD);
  zero_f<<<1, 64, 0, stream>>>(smallBuf, 32);

  const dim3 gProj((unsigned)(M / 64), D / 128, 1), bl(256);
  const dim3 gSim(T / 64, T / 128, B);
  const dim3 gCtx(T / 16, D / 64, B), bw(32);

  for (int tick = 0; tick < 3; ++tick) {
    const float* zCur = (tick == 0) ? zIn : zState;
    float* zNext = (tick == 2) ? (float*)d_out : zState;
    zero_f<<<1, 32, 0, stream>>>(confSum, 8);

    // h = LN(z @ in_w^T + in_b)
    gemm_wmma<<<gProj, bl, 0, stream>>>(z16, w16[0], in_b, fbufA, nullptr, nullptr,
                                        (int)M, D, D, 1.0f, 0, 0, 0, 0);
    ln_kernel<<<(unsigned)(M / 8), 256, 0, stream>>>(fbufA, ln_g, ln_b, h16);

    // Q, K, V (V written transposed for the P@V GEMM)
    gemm_wmma<<<gProj, bl, 0, stream>>>(h16, w16[1], q_b, nullptr, q16, nullptr,
                                        (int)M, D, D, 1.0f, 0, 0, 0, 0);
    gemm_wmma<<<gProj, bl, 0, stream>>>(h16, w16[2], k_b, nullptr, k16, nullptr,
                                        (int)M, D, D, 1.0f, 0, 0, 0, 0);
    gemm_wmma<<<gProj, bl, 0, stream>>>(h16, w16[3], v_b, nullptr, nullptr, vt16,
                                        (int)M, D, D, 1.0f, 0, 0, 0, 0);

    // sim = Q @ K^T * 1/sqrt(D), materialized f16 (fits in L2)
    gemm_wmma<<<gSim, bl, 0, stream>>>(q16, k16, nullptr, nullptr, sim16, nullptr,
                                       T, T, D, inv_sqrt_d, 0, (long long)T * D,
                                       (long long)T * D, (long long)T * T);

    // confidence -> eff_p, then sharpened row stats
    attn_stats<<<(unsigned)(M / 8), 256, 0, stream>>>(sim16, nullptr, confSum,
                                                      nullptr, nullptr, 0);
    conf_kernel<<<1, 32, 0, stream>>>(confSum, effp, effpSum);
    attn_stats<<<(unsigned)(M / 8), 256, 0, stream>>>(sim16, effp, nullptr, m2, z2,
                                                      1);

    // context = softmax(sim * eff_p) @ V
    attn_ctx<<<gCtx, bw, 0, stream>>>(sim16, vt16, effp, m2, z2, ctx16);

    // target = ctx @ o_w^T + o_b ; z_pred = relu(z @ p1^T + b) @ p2^T + b
    gemm_wmma<<<gProj, bl, 0, stream>>>(ctx16, w16[4], o_b, fbufA, nullptr, nullptr,
                                        (int)M, D, D, 1.0f, 0, 0, 0, 0);
    gemm_wmma<<<gProj, bl, 0, stream>>>(z16, w16[5], p1_b, nullptr, r116, nullptr,
                                        (int)M, D, D, 1.0f, 1, 0, 0, 0);
    gemm_wmma<<<gProj, bl, 0, stream>>>(r116, w16[6], p2_b, fbufB, nullptr, nullptr,
                                        (int)M, D, D, 1.0f, 0, 0, 0, 0);

    // z_next = z + (target - z)/(tau+1e-6); accumulate MSE; refresh z16
    update_kernel<<<4096, 256, 0, stream>>>(zCur, fbufA, fbufB, log_tau, zNext,
                                            z16, lossSum, MD);
  }

  finalize<<<1, 1, 0, stream>>>(lossSum, effpSum, (float*)d_out, MD);
}
